// CausalSelfAttention_84473416777919
// MI455X (gfx1250) — matrix operand order repaired and checked by reading, not compiled, not run on hardware
//
#include <hip/hip_runtime.h>

typedef __bf16 bf16;
typedef __attribute__((ext_vector_type(16))) __bf16 bf16x16;
typedef __attribute__((ext_vector_type(8)))  __bf16 bf16x8;
typedef __attribute__((ext_vector_type(8)))  float  f32x8;

union Frag16 { bf16x16 f; bf16x8 h[2]; };

// ---------------------------------------------------------------- cvt fp32 -> bf16
__global__ void cvt_f32_bf16(const float* __restrict__ in, bf16* __restrict__ out, int n) {
    int i = blockIdx.x * blockDim.x + threadIdx.x;
    if (i < n) out[i] = (bf16)in[i];
}

// ---------------------------------------------------------------- tiled bf16 WMMA GEMM
// C[M,N] (fp32, row-major) = A[M,K] (bf16 rm) * B[K,N] (bf16 rm)
// 256 threads = 8 waves; 128x128 C tile / WG; wave = 64(M)x32(N); K chunk 32.
// Register ping-pong: next chunk is fetched into VGPRs while WMMAs consume LDS.
template<int N, int K>
__global__ __launch_bounds__(256)
void gemm_bf16_wmma(const bf16* __restrict__ A, const bf16* __restrict__ B,
                    float* __restrict__ C) {
    __shared__ __align__(16) bf16 Al[128][40];   // [row][k], padded
    __shared__ __align__(16) bf16 Bt[128][40];   // [col][k], transposed

    const int tid  = threadIdx.x;
    const int lane = tid & 31;
    const int w    = tid >> 5;
    const int wm   = w & 1;
    const int wn   = w >> 1;
    const int l16  = lane & 15;
    const int hiH  = lane >> 4;
    const int akb  = hiH ? 16 : 0;
    const int bkb  = hiH ? 16 : 0;
    const int mBase = blockIdx.y * 128;
    const int nBase = blockIdx.x * 128;

    // per-thread tile-load coordinates (two 8-element vectors each)
    const int e0 = tid * 8,        e1 = tid * 8 + 2048;
    const int ar0 = e0 >> 5, ac0 = e0 & 31;
    const int ar1 = e1 >> 5, ac1 = e1 & 31;
    const int br0 = e0 >> 7, bc0 = e0 & 127;
    const int br1 = e1 >> 7, bc1 = e1 & 127;
    const bf16* aP0 = A + (size_t)(mBase + ar0) * K + ac0;
    const bf16* aP1 = A + (size_t)(mBase + ar1) * K + ac1;
    const bf16* bP0 = B + (size_t)br0 * N + nBase + bc0;
    const bf16* bP1 = B + (size_t)br1 * N + nBase + bc1;

    f32x8 acc[4][2];
    for (int mt = 0; mt < 4; ++mt)
        for (int nt = 0; nt < 2; ++nt)
            for (int j = 0; j < 8; ++j) acc[mt][nt][j] = 0.0f;

    // prologue: chunk 0 into registers
    bf16x8 ra0 = *(const bf16x8*)aP0;
    bf16x8 ra1 = *(const bf16x8*)aP1;
    bf16x8 rb0 = *(const bf16x8*)bP0;
    bf16x8 rb1 = *(const bf16x8*)bP1;

    for (int kk = 0; kk < K; kk += 32) {
        // commit current chunk to LDS
        *(bf16x8*)&Al[ar0][ac0] = ra0;
        *(bf16x8*)&Al[ar1][ac1] = ra1;
        #pragma unroll
        for (int j = 0; j < 8; ++j) Bt[bc0 + j][br0] = rb0[j];
        #pragma unroll
        for (int j = 0; j < 8; ++j) Bt[bc1 + j][br1] = rb1[j];
        __syncthreads();

        // prefetch next chunk into registers (overlaps with WMMA below)
        if (kk + 32 < K) {
            aP0 += 32; aP1 += 32;
            bP0 += (size_t)32 * N; bP1 += (size_t)32 * N;
            ra0 = *(const bf16x8*)aP0;
            ra1 = *(const bf16x8*)aP1;
            rb0 = *(const bf16x8*)bP0;
            rb1 = *(const bf16x8*)bP1;
            if (kk + 64 < K) {                       // warm L2 one chunk further
                __builtin_prefetch(aP0 + 32, 0, 0);
                __builtin_prefetch(bP0 + (size_t)32 * N, 0, 0);
            }
        }

        Frag16 af[4], bfr[2];
        #pragma unroll
        for (int mt = 0; mt < 4; ++mt) {
            int row = wm * 64 + mt * 16 + l16;
            af[mt].h[0] = *(const bf16x8*)&Al[row][akb];
            af[mt].h[1] = *(const bf16x8*)&Al[row][akb + 8];
        }
        #pragma unroll
        for (int nt = 0; nt < 2; ++nt) {
            int col = wn * 32 + nt * 16 + l16;
            bfr[nt].h[0] = *(const bf16x8*)&Bt[col][bkb];
            bfr[nt].h[1] = *(const bf16x8*)&Bt[col][bkb + 8];
        }
        #pragma unroll
        for (int mt = 0; mt < 4; ++mt)
            #pragma unroll
            for (int nt = 0; nt < 2; ++nt)
                acc[mt][nt] = __builtin_amdgcn_wmma_f32_16x16x32_bf16(
                    false, af[mt].f, false, bfr[nt].f, (short)0, acc[mt][nt], false, false);
        __syncthreads();
    }

    // epilogue: one base pointer, compile-time immediate offsets
    float* Cp = C + (size_t)(mBase + wm * 64 + 8 * hiH) * N + nBase + wn * 32 + l16;
    #pragma unroll
    for (int mt = 0; mt < 4; ++mt)
        #pragma unroll
        for (int nt = 0; nt < 2; ++nt)
            #pragma unroll
            for (int j = 0; j < 8; ++j)
                Cp[(size_t)(mt * 16 + j) * N + nt * 16] = acc[mt][nt][j];
}

// ---------------------------------------------------------------- RoPE + repack
// qkv fp32 [B,L,3*D] -> q/k/v bf16 [B,N,L,H]; q pre-scaled by 1/sqrt(H)
__global__ void rope_pack(const float* __restrict__ qkv, const float* __restrict__ rope,
                          bf16* __restrict__ qb, bf16* __restrict__ kb, bf16* __restrict__ vb) {
    int i  = blockIdx.x * blockDim.x + threadIdx.x;   // B*L*N*(H/2) = 2^21
    int hh = i & 31;
    int n  = (i >> 5) & 15;
    int l  = (i >> 9) & 2047;
    int b  = i >> 20;
    size_t base = ((size_t)b * 2048 + l) * 3072 + n * 64 + 2 * hh;
    float q0 = qkv[base],        q1 = qkv[base + 1];
    float k0 = qkv[base + 1024], k1 = qkv[base + 1025];
    float v0 = qkv[base + 2048], v1 = qkv[base + 2049];
    float c = rope[(size_t)l * 64 + 2 * hh];
    float s = rope[(size_t)l * 64 + 2 * hh + 1];
    size_t ob = (((size_t)b * 16 + n) * 2048 + l) * 64 + 2 * hh;
    const float sc = 0.125f;   // 1/sqrt(64)
    qb[ob]     = (bf16)((q0 * c - q1 * s) * sc);
    qb[ob + 1] = (bf16)((q0 * s + q1 * c) * sc);
    kb[ob]     = (bf16)(k0 * c - k1 * s);
    kb[ob + 1] = (bf16)(k0 * s + k1 * c);
    vb[ob]     = (bf16)v0;
    vb[ob + 1] = (bf16)v1;
}

// ---------------------------------------------------------------- flash attention
// 4 waves/WG; WG = 64 consecutive queries of one head; wave owns 16 queries.
// K/V tiles register-prefetched while current tile is consumed. out bf16 [B,L,N*H].
__global__ __launch_bounds__(128)
void flash_attn(const bf16* __restrict__ qb, const bf16* __restrict__ kb,
                const bf16* __restrict__ vb, bf16* __restrict__ ob) {
    __shared__ __align__(16) bf16 Kl[32][72];       // [key][h]
    __shared__ __align__(16) bf16 VT[64][40];       // [h][key]
    __shared__ __align__(16) bf16 Pl[4][16][40];    // per-wave P staging

    const int tid  = threadIdx.x;
    const int lane = tid & 31;
    const int w    = tid >> 5;
    const int l16  = lane & 15;
    const int hiH  = lane >> 4;
    const int akb  = hiH ? 16 : 0;
    const int bkb  = hiH ? 16 : 0;

    const int LT   = 2048 / 64;
    int head = blockIdx.x / LT;               // b*16 + n
    int q64  = (blockIdx.x % LT) * 64;
    size_t hb = (size_t)head * 2048 * 64;
    int q0 = q64 + w * 16;

    Frag16 qf[2];
    {
        const bf16* qp = qb + hb + (size_t)(q0 + l16) * 64;
        qf[0].h[0] = *(const bf16x8*)(qp + akb);
        qf[0].h[1] = *(const bf16x8*)(qp + akb + 8);
        qf[1].h[0] = *(const bf16x8*)(qp + 32 + akb);
        qf[1].h[1] = *(const bf16x8*)(qp + 32 + akb + 8);
    }

    f32x8 o[4];
    for (int ht = 0; ht < 4; ++ht)
        for (int j = 0; j < 8; ++j) o[ht][j] = 0.0f;
    float m[8], lsum[8];
    for (int j = 0; j < 8; ++j) { m[j] = -1e30f; lsum[j] = 0.0f; }

    // per-thread K/V tile-load coordinates
    const int e0 = tid * 8,      e1 = tid * 8 + 1024;
    const int rr0 = e0 >> 6, cc0 = e0 & 63;
    const int rr1 = e1 >> 6, cc1 = e1 & 63;
    const bf16* kP0 = kb + hb + (size_t)rr0 * 64 + cc0;
    const bf16* kP1 = kb + hb + (size_t)rr1 * 64 + cc1;
    const bf16* vP0 = vb + hb + (size_t)rr0 * 64 + cc0;
    const bf16* vP1 = vb + hb + (size_t)rr1 * 64 + cc1;

    int nkt = q64 / 32 + 2;
    bf16x8 rk0 = *(const bf16x8*)kP0;
    bf16x8 rk1 = *(const bf16x8*)kP1;
    bf16x8 rv0 = *(const bf16x8*)vP0;
    bf16x8 rv1 = *(const bf16x8*)vP1;

    for (int kt = 0; kt < nkt; ++kt) {
        int kk = kt * 32;
        // commit tile kt to LDS
        *(bf16x8*)&Kl[rr0][cc0] = rk0;
        *(bf16x8*)&Kl[rr1][cc1] = rk1;
        #pragma unroll
        for (int j = 0; j < 8; ++j) VT[cc0 + j][rr0] = rv0[j];
        #pragma unroll
        for (int j = 0; j < 8; ++j) VT[cc1 + j][rr1] = rv1[j];
        __syncthreads();

        // register-prefetch tile kt+1
        if (kt + 1 < nkt) {
            kP0 += 32 * 64; kP1 += 32 * 64;
            vP0 += 32 * 64; vP1 += 32 * 64;
            rk0 = *(const bf16x8*)kP0;
            rk1 = *(const bf16x8*)kP1;
            rv0 = *(const bf16x8*)vP0;
            rv1 = *(const bf16x8*)vP1;
        }

        if (kk <= q0 + 15) {
            f32x8 s[2];
            #pragma unroll
            for (int st = 0; st < 2; ++st) {
                for (int j = 0; j < 8; ++j) s[st][j] = 0.0f;
                Frag16 kf0, kf1;
                int krow = st * 16 + l16;
                kf0.h[0] = *(const bf16x8*)&Kl[krow][bkb];
                kf0.h[1] = *(const bf16x8*)&Kl[krow][bkb + 8];
                kf1.h[0] = *(const bf16x8*)&Kl[krow][32 + bkb];
                kf1.h[1] = *(const bf16x8*)&Kl[krow][32 + bkb + 8];
                s[st] = __builtin_amdgcn_wmma_f32_16x16x32_bf16(
                    false, qf[0].f, false, kf0.f, (short)0, s[st], false, false);
                s[st] = __builtin_amdgcn_wmma_f32_16x16x32_bf16(
                    false, qf[1].f, false, kf1.f, (short)0, s[st], false, false);
            }
            #pragma unroll
            for (int j = 0; j < 8; ++j) {
                int row = q0 + j + 8 * hiH;
                float v0 = (kk + l16      <= row) ? s[0][j] : -1e30f;
                float v1 = (kk + 16 + l16 <= row) ? s[1][j] : -1e30f;
                float rm = fmaxf(v0, v1);
                rm = fmaxf(rm, __shfl_xor(rm, 1));
                rm = fmaxf(rm, __shfl_xor(rm, 2));
                rm = fmaxf(rm, __shfl_xor(rm, 4));
                rm = fmaxf(rm, __shfl_xor(rm, 8));
                float nm = fmaxf(m[j], rm);
                float alpha = __expf(m[j] - nm);
                m[j] = nm;
                float p0 = __expf(v0 - nm);
                float p1 = __expf(v1 - nm);
                float rs = p0 + p1;
                rs += __shfl_xor(rs, 1);
                rs += __shfl_xor(rs, 2);
                rs += __shfl_xor(rs, 4);
                rs += __shfl_xor(rs, 8);
                lsum[j] = lsum[j] * alpha + rs;
                #pragma unroll
                for (int ht = 0; ht < 4; ++ht) o[ht][j] *= alpha;
                int rl = j + 8 * hiH;
                Pl[w][rl][l16]      = (bf16)p0;
                Pl[w][rl][16 + l16] = (bf16)p1;
            }
            Frag16 pf;
            pf.h[0] = *(const bf16x8*)&Pl[w][l16][akb];
            pf.h[1] = *(const bf16x8*)&Pl[w][l16][akb + 8];
            #pragma unroll
            for (int ht = 0; ht < 4; ++ht) {
                Frag16 vf;
                int hc = ht * 16 + l16;
                vf.h[0] = *(const bf16x8*)&VT[hc][bkb];
                vf.h[1] = *(const bf16x8*)&VT[hc][bkb + 8];
                o[ht] = __builtin_amdgcn_wmma_f32_16x16x32_bf16(
                    false, pf.f, false, vf.f, (short)0, o[ht], false, false);
            }
        }
        __syncthreads();
    }

    int b = head >> 4, n = head & 15;
    #pragma unroll
    for (int j = 0; j < 8; ++j) {
        float inv = 1.0f / lsum[j];
        int row = q0 + j + 8 * hiH;
        size_t rb = ((size_t)b * 2048 + row) * 1024 + n * 64;
        #pragma unroll
        for (int ht = 0; ht < 4; ++ht)
            ob[rb + ht * 16 + l16] = (bf16)(o[ht][j] * inv);
    }
}

// ---------------------------------------------------------------- launch
extern "C" void kernel_launch(void* const* d_in, const int* in_sizes, int n_in,
                              void* d_out, int out_size, void* d_ws, size_t ws_size,
                              hipStream_t stream) {
    const float* x      = (const float*)d_in[0];
    const float* rope   = (const float*)d_in[1];
    // d_in[2] = mask (bool) — causal mask applied analytically
    const float* w_qkv  = (const float*)d_in[3];
    const float* w_proj = (const float*)d_in[4];
    float* out = (float*)d_out;

    const int B = 2, L = 2048, D = 1024;
    const size_t nX    = (size_t)B * L * D;
    const size_t nWqkv = (size_t)D * 3 * D;
    const size_t nWprj = (size_t)D * D;
    const size_t nQkv  = (size_t)B * L * 3 * D;
    const size_t nHead = (size_t)B * 16 * L * 64;

    char* ws = (char*)d_ws;
    size_t off = 0;
    auto alloc = [&](size_t bytes) -> void* {
        void* p = ws + off;
        off = (off + bytes + 255) & ~(size_t)255;
        return p;
    };
    bf16*  xb     = (bf16*)alloc(nX * 2);
    bf16*  wqkvb  = (bf16*)alloc(nWqkv * 2);
    bf16*  wprojb = (bf16*)alloc(nWprj * 2);
    float* qkvf   = (float*)alloc(nQkv * 4);
    bf16*  qbuf   = (bf16*)alloc(nHead * 2);
    bf16*  kbuf   = (bf16*)alloc(nHead * 2);
    bf16*  vbuf   = (bf16*)alloc(nHead * 2);
    bf16*  attnb  = (bf16*)alloc(nX * 2);
    (void)ws_size; (void)in_sizes; (void)n_in; (void)out_size;

    cvt_f32_bf16<<<(int)(nX    / 256), 256, 0, stream>>>(x,      xb,     (int)nX);
    cvt_f32_bf16<<<(int)(nWqkv / 256), 256, 0, stream>>>(w_qkv,  wqkvb,  (int)nWqkv);
    cvt_f32_bf16<<<(int)(nWprj / 256), 256, 0, stream>>>(w_proj, wprojb, (int)nWprj);

    gemm_bf16_wmma<3072, 1024><<<dim3(3072 / 128, 4096 / 128), 256, 0, stream>>>(
        xb, wqkvb, qkvf);

    rope_pack<<<(int)((size_t)B * L * 16 * 32 / 256), 256, 0, stream>>>(
        qkvf, rope, qbuf, kbuf, vbuf);

    flash_attn<<<32 * (2048 / 64), 128, 0, stream>>>(qbuf, kbuf, vbuf, attnb);

    gemm_bf16_wmma<1024, 1024><<<dim3(1024 / 128, 4096 / 128), 256, 0, stream>>>(
        attnb, wprojb, out);
}
